// GroundingDINOCriterion_62302795596298
// MI455X (gfx1250) — compile-verified, matrix operand order checked by reading, not run on hardware
//
#include <hip/hip_runtime.h>
#include <hip/hip_bf16.h>
#include <math.h>

// Problem constants (from reference)
#define BB 128
#define QQ 900
#define TT 256
#define GG 20
#define EPSV 1e-6f

#define NFOCAL_BLOCKS 1024

// Workspace layout (float slots)
#define WS_FOCAL 0            // NFOCAL_BLOCKS floats
#define WS_CORR  1024         // BB*GG floats
#define WS_LB    3584         // BB floats
#define WS_LG    3712         // BB floats
#define WS_MATCH 3840         // BB*GG ints

typedef __attribute__((ext_vector_type(16))) _Float16 v16h;
typedef __attribute__((ext_vector_type(8)))  float    v8f;

// ---------------------------------------------------------------------------
// Wave-level (32-lane) sum reduction via one V_WMMA_F32_16X16X32_F16.
// Each lane contributes its f32 partial as a double-f16 split (hi + residual)
// placed in a[0], a[1]; all other A entries zero. B = ones. Since D[m][n] =
// rowsum_m(A), summing D over all m for any fixed column n gives total(A),
// independent of the exact A lane->(m,k) mapping. Accumulation is f32, so the
// only error is the ~2^-21 relative error of the hi/lo split.
// EXEC must be all ones: call only from convergent code.
// ---------------------------------------------------------------------------
__device__ __forceinline__ float wave_reduce_wmma(float x) {
  v16h a = {};
  _Float16 hi = (_Float16)x;
  a[0] = hi;
  a[1] = (_Float16)(x - (float)hi);
  v16h ones;
#pragma unroll
  for (int i = 0; i < 16; ++i) ones[i] = (_Float16)1.0f;
  v8f c = {};
  c = __builtin_amdgcn_wmma_f32_16x16x32_f16(
      /*neg_a=*/false, a, /*neg_b=*/false, ones,
      /*c_mod=*/(short)0, c, /*reuse_a=*/false, /*reuse_b=*/false);
  float s = c[0] + c[1] + c[2] + c[3] + c[4] + c[5] + c[6] + c[7];
  // s at lane L<16 = sum_{m=0..7} D[m][L]; at lane L+16 = sum_{m=8..15} D[m][L]
  return __shfl(s, 0, 32) + __shfl(s, 16, 32);
}

// focal loss element, target = 0:  0.75 * softplus(x) * sigmoid(x)^2
__device__ __forceinline__ float focal0(float x) {
  float e = __expf(-fabsf(x));          // e^{-|x|} in (0,1]
  float r = 1.0f / (1.0f + e);
  float p = (x >= 0.0f) ? r : e * r;    // sigmoid(x)
  float sp = fmaxf(x, 0.0f) + log1pf(e); // softplus(x) = CE with target 0
  return 0.75f * sp * p * p;
}

// ---------------------------------------------------------------------------
// Kernel 1: per-batch greedy matching + bbox/giou losses.
// One 256-thread block per batch. Pred boxes cached in registers (4/thread),
// targets in LDS. 20 sequential argmax tree-reductions with alive flags
// (exactly equivalent to the reference's row-kill scheme since IoU >= 0).
// Tie-break: lowest index (jnp.argmax first-occurrence semantics).
// ---------------------------------------------------------------------------
__global__ __launch_bounds__(256) void match_kernel(
    const float* __restrict__ pred_boxes,  // [B, Q, 4] cxcywh
    const float* __restrict__ tgt_boxes,   // [B, G, 4] cxcywh
    float* __restrict__ ws) {
  const int b = blockIdx.x;
  const int tid = threadIdx.x;

  __shared__ float4 s_tgt[GG];   // xyxy
  __shared__ float  s_tarea[GG];
  __shared__ float  s_val[256];
  __shared__ int    s_idx[256];
  __shared__ int    s_match[GG];
  __shared__ float  s_lb[GG];
  __shared__ float  s_lgv[GG];

  if (tid < GG) {
    const float4 tb = *(const float4*)(tgt_boxes + ((size_t)b * GG + tid) * 4);
    float4 xy;
    xy.x = tb.x - 0.5f * tb.z; xy.y = tb.y - 0.5f * tb.w;
    xy.z = tb.x + 0.5f * tb.z; xy.w = tb.y + 0.5f * tb.w;
    s_tgt[tid] = xy;
    s_tarea[tid] = (xy.z - xy.x) * (xy.w - xy.y);
  }
  __syncthreads();

  float4 pxy[4];
  float  parea[4];
  bool   alive[4];
#pragma unroll
  for (int k = 0; k < 4; ++k) {
    const int q = tid + k * 256;
    alive[k] = (q < QQ);
    if (alive[k]) {
      const float4 pb = *(const float4*)(pred_boxes + ((size_t)b * QQ + q) * 4);
      float4 xy;
      xy.x = pb.x - 0.5f * pb.z; xy.y = pb.y - 0.5f * pb.w;
      xy.z = pb.x + 0.5f * pb.z; xy.w = pb.y + 0.5f * pb.w;
      pxy[k] = xy;
      parea[k] = (xy.z - xy.x) * (xy.w - xy.y);
    } else {
      pxy[k] = make_float4(0.f, 0.f, 0.f, 0.f);
      parea[k] = 0.f;
    }
  }

  for (int g = 0; g < GG; ++g) {
    const float4 t = s_tgt[g];
    const float ta = s_tarea[g];
    float best = -1e30f;
    int   bi = 0x7fffffff;
#pragma unroll
    for (int k = 0; k < 4; ++k) {
      if (alive[k]) {
        float lx = fmaxf(pxy[k].x, t.x), ly = fmaxf(pxy[k].y, t.y);
        float rx = fminf(pxy[k].z, t.z), ry = fminf(pxy[k].w, t.w);
        float w = fmaxf(rx - lx, 0.0f), h = fmaxf(ry - ly, 0.0f);
        float inter = w * h;
        float iou = inter / (parea[k] + ta - inter);
        if (iou > best) { best = iou; bi = tid + k * 256; }  // ascending q: ties keep lowest
      }
    }
    s_val[tid] = best;
    s_idx[tid] = bi;
    __syncthreads();
    for (int off = 128; off > 0; off >>= 1) {
      if (tid < off) {
        float v2 = s_val[tid + off];
        int   i2 = s_idx[tid + off];
        if (v2 > s_val[tid] || (v2 == s_val[tid] && i2 < s_idx[tid])) {
          s_val[tid] = v2; s_idx[tid] = i2;
        }
      }
      __syncthreads();
    }
    if (tid == 0) s_match[g] = s_idx[0];
    __syncthreads();
    const int m = s_match[g];
#pragma unroll
    for (int k = 0; k < 4; ++k)
      if (tid + k * 256 == m) alive[k] = false;
  }

  // losses over the 20 matched pairs
  if (tid < GG) {
    const int g = tid;
    const int q = s_match[g];
    const float4 mb = *(const float4*)(pred_boxes + ((size_t)b * QQ + q) * 4);
    const float4 tb = *(const float4*)(tgt_boxes + ((size_t)b * GG + g) * 4);
    const float l1 = fabsf(mb.x - tb.x) + fabsf(mb.y - tb.y) +
                     fabsf(mb.z - tb.z) + fabsf(mb.w - tb.w);

    // matched xyxy -> ensure_valid
    float mx1 = mb.x - 0.5f * mb.z, my1 = mb.y - 0.5f * mb.w;
    float mx2 = mb.x + 0.5f * mb.z, my2 = mb.y + 0.5f * mb.w;
    mx2 = fmaxf(mx2, mx1 + EPSV); my2 = fmaxf(my2, my1 + EPSV);
    mx1 = fminf(fmaxf(mx1, 0.f), 1.f); my1 = fminf(fmaxf(my1, 0.f), 1.f);
    mx2 = fminf(fmaxf(mx2, 0.f), 1.f); my2 = fminf(fmaxf(my2, 0.f), 1.f);
    // target xyxy -> ensure_valid
    float4 txy = s_tgt[g];
    float tx1 = txy.x, ty1 = txy.y, tx2 = txy.z, ty2 = txy.w;
    tx2 = fmaxf(tx2, tx1 + EPSV); ty2 = fmaxf(ty2, ty1 + EPSV);
    tx1 = fminf(fmaxf(tx1, 0.f), 1.f); ty1 = fminf(fmaxf(ty1, 0.f), 1.f);
    tx2 = fminf(fmaxf(tx2, 0.f), 1.f); ty2 = fminf(fmaxf(ty2, 0.f), 1.f);

    const float a1 = (mx2 - mx1) * (my2 - my1);
    const float a2 = (tx2 - tx1) * (ty2 - ty1);
    const float lx = fmaxf(mx1, tx1), ly = fmaxf(my1, ty1);
    const float rx = fminf(mx2, tx2), ry = fminf(my2, ty2);
    const float w = fmaxf(rx - lx, 0.f), h = fmaxf(ry - ly, 0.f);
    const float inter = w * h;
    const float uni = a1 + a2 - inter;
    const float iou = inter / uni;
    const float cx1 = fminf(mx1, tx1), cy1 = fminf(my1, ty1);
    const float cx2 = fmaxf(mx2, tx2), cy2 = fmaxf(my2, ty2);
    const float ac = fmaxf(cx2 - cx1, 0.f) * fmaxf(cy2 - cy1, 0.f);
    const float giou = iou - (ac - uni) / ac;

    s_lb[g] = l1;
    s_lgv[g] = 1.0f - giou;
    ((int*)ws)[WS_MATCH + b * GG + g] = q;
  }
  __syncthreads();
  if (tid == 0) {
    float lb = 0.f, lg = 0.f;
    for (int g = 0; g < GG; ++g) { lb += s_lb[g]; lg += s_lgv[g]; }
    ws[WS_LB + b] = lb;
    ws[WS_LG + b] = lg;
  }
}

// ---------------------------------------------------------------------------
// Kernel 2: target=0 focal loss over all 29.5M logits (memory-bound stream).
// float4 loads, grid-stride, per-wave WMMA reduction, per-block partial to ws.
// ---------------------------------------------------------------------------
__global__ __launch_bounds__(256) void focal_kernel(
    const float4* __restrict__ logits4, float* __restrict__ ws) {
  const long long N4 = (long long)BB * QQ * TT / 4;
  const long long stride = (long long)gridDim.x * 256;
  long long i = (long long)blockIdx.x * 256 + threadIdx.x;

  float acc = 0.0f;
  for (; i < N4; i += stride) {
    __builtin_prefetch(logits4 + i + stride, 0, 0);  // global_prefetch_b8
    const float4 v = logits4[i];
    acc += focal0(v.x) + focal0(v.y) + focal0(v.z) + focal0(v.w);
  }

  // convergent here: every wave has full EXEC for the WMMA reduction
  const float wtot = wave_reduce_wmma(acc);

  __shared__ float s_w[8];
  const int lane = threadIdx.x & 31;
  const int wid = threadIdx.x >> 5;
  if (lane == 0) s_w[wid] = wtot;
  __syncthreads();
  if (threadIdx.x == 0) {
    float bsum = 0.f;
#pragma unroll
    for (int w = 0; w < 8; ++w) bsum += s_w[w];
    ws[WS_FOCAL + blockIdx.x] = bsum;
  }
}

// ---------------------------------------------------------------------------
// Kernel 3: corrections where target=1 (matched rows, inside token span):
// add focal(x,1) - focal(x,0) for each such position. One thread per (b,g).
// ---------------------------------------------------------------------------
__global__ __launch_bounds__(256) void corr_kernel(
    const float* __restrict__ logits,   // [B, Q, T]
    const int* __restrict__ spans,      // [B, G, 2]
    float* __restrict__ ws) {
  const int idx = blockIdx.x * 256 + threadIdx.x;
  if (idx >= BB * GG) return;
  const int b = idx / GG;
  const int q = ((const int*)ws)[WS_MATCH + idx];
  const int s = spans[idx * 2 + 0];
  const int e = spans[idx * 2 + 1];
  const float* row = logits + ((size_t)b * QQ + q) * TT;
  float acc = 0.0f;
  for (int t = s; t < e; ++t) {
    const float x = row[t];
    const float ex = __expf(-fabsf(x));
    const float r = 1.0f / (1.0f + ex);
    const float p = (x >= 0.0f) ? r : ex * r;
    const float l1p = log1pf(ex);
    const float f0 = 0.75f * (fmaxf(x, 0.0f) + l1p) * p * p;
    const float f1 = 0.25f * (fmaxf(x, 0.0f) - x + l1p) * (1.0f - p) * (1.0f - p);
    acc += f1 - f0;
  }
  ws[WS_CORR + idx] = acc;
}

// ---------------------------------------------------------------------------
// Kernel 4: combine everything, apply coefficients and normalizations.
// ---------------------------------------------------------------------------
__global__ __launch_bounds__(256) void finalize_kernel(
    const float* __restrict__ ws, float* __restrict__ out) {
  const int tid = threadIdx.x;
  float aF = 0.f, aC = 0.f, aB = 0.f, aG = 0.f;
  for (int i = tid; i < NFOCAL_BLOCKS; i += 256) aF += ws[WS_FOCAL + i];
  for (int i = tid; i < BB * GG; i += 256)       aC += ws[WS_CORR + i];
  for (int i = tid; i < BB; i += 256) { aB += ws[WS_LB + i]; aG += ws[WS_LG + i]; }

  __shared__ float4 red[256];
  red[tid] = make_float4(aF, aC, aB, aG);
  __syncthreads();
  for (int off = 128; off > 0; off >>= 1) {
    if (tid < off) {
      float4 o = red[tid + off];
      float4 m = red[tid];
      m.x += o.x; m.y += o.y; m.z += o.z; m.w += o.w;
      red[tid] = m;
    }
    __syncthreads();
  }
  if (tid == 0) {
    const float4 r = red[0];
    const float inv_nb = 1.0f / (float)(BB * GG);
    const float lb = 5.0f * r.z * inv_nb;
    const float lg = 2.0f * r.w * inv_nb;
    const float lc = 1.0f * ((r.x + r.y) / (float)(QQ * TT)) * inv_nb;
    out[0] = lb;
    out[1] = lg;
    out[2] = lc;
    out[3] = lb + lg + lc;
  }
}

extern "C" void kernel_launch(void* const* d_in, const int* in_sizes, int n_in,
                              void* d_out, int out_size, void* d_ws, size_t ws_size,
                              hipStream_t stream) {
  (void)in_sizes; (void)n_in; (void)out_size; (void)ws_size;
  const float* pred_logits = (const float*)d_in[0];
  const float* pred_boxes  = (const float*)d_in[1];
  const float* tgt_boxes   = (const float*)d_in[2];
  const int*   tokens      = (const int*)d_in[3];
  float* ws = (float*)d_ws;
  float* out = (float*)d_out;

  hipLaunchKernelGGL(match_kernel, dim3(BB), dim3(256), 0, stream,
                     pred_boxes, tgt_boxes, ws);
  hipLaunchKernelGGL(focal_kernel, dim3(NFOCAL_BLOCKS), dim3(256), 0, stream,
                     (const float4*)pred_logits, ws);
  hipLaunchKernelGGL(corr_kernel, dim3((BB * GG + 255) / 256), dim3(256), 0, stream,
                     pred_logits, tokens, ws);
  hipLaunchKernelGGL(finalize_kernel, dim3(1), dim3(256), 0, stream, ws, out);
}
